// _SupConLossWithNorm_65420941852911
// MI455X (gfx1250) — compile-verified, matrix operand order checked by reading, not run on hardware
//
#include <hip/hip_runtime.h>

typedef __attribute__((ext_vector_type(16))) _Float16 v16h;
typedef __attribute__((ext_vector_type(8)))  _Float16 v8h;
typedef __attribute__((ext_vector_type(4)))  _Float16 v4h;
typedef __attribute__((ext_vector_type(8)))  float    v8f;
typedef __attribute__((ext_vector_type(4)))  unsigned int uint32x4;
typedef __attribute__((ext_vector_type(8)))  int      int32x8;
typedef __attribute__((ext_vector_type(4)))  int      int32x4;

#define D_DIM 128
#define NORM_EPS 1e-12f
#define INV_TEMP 10.0f
#define SHIFT 10.0f   // fixed logsumexp shift: rows unit-norm => s <= ~1/T
#define SMALL_VAL 1.1754943508222875e-38f

// LDS panel: rows of 256B padded to 272B by TDM (4 DWORDs per 64 DWORDs) so
// ds_load_b128 fragment reads rotate 4 banks/row -> conflict-free.
#define ROW_BYTES   256
#define LDS_ROW_B   272
#define STAGE_ROWS  32                       // 32 columns of the Gram per stage
#define PANEL_BYTES (STAGE_ROWS * LDS_ROW_B) // 8704
#define WAVES_PB    4                        // 4 waves/block share each panel

#if defined(__has_builtin)
#if __has_builtin(__builtin_amdgcn_tensor_load_to_lds) && \
    __has_builtin(__builtin_amdgcn_s_wait_tensorcnt)
#define HAVE_TDM 1
#endif
#endif

union V16U { v16h v; v8h h[2]; };

// ---------------------------------------------------------------------------
// Kernel 1: row L2-normalize fp32 -> f16.  One wave (32 lanes x float4) per row.
// ---------------------------------------------------------------------------
__global__ __launch_bounds__(256) void supcon_normalize(
    const float* __restrict__ emb, _Float16* __restrict__ xh, int N) {
  const int lane = threadIdx.x & 31;
  const int wave = threadIdx.x >> 5;
  const int row  = blockIdx.x * 8 + wave;
  if (row >= N) return;

  const float4 v = ((const float4*)(emb + (size_t)row * D_DIM))[lane];
  float ss = v.x * v.x + v.y * v.y + v.z * v.z + v.w * v.w;
#pragma unroll
  for (int m = 16; m >= 1; m >>= 1) ss += __shfl_xor(ss, m, 32);
  const float inv = 1.0f / fmaxf(sqrtf(ss), NORM_EPS);

  v4h o;
  o[0] = (_Float16)(v.x * inv);
  o[1] = (_Float16)(v.y * inv);
  o[2] = (_Float16)(v.z * inv);
  o[3] = (_Float16)(v.w * inv);
  ((v4h*)(xh + (size_t)row * D_DIM))[lane] = o;
}

// Load one 16x32 f16 WMMA fragment chunk (K base = 32*c) from a row-major
// panel; base_row already points at this lane's row (lmod).
// ISA layout: lanes 0-15 K 0-7 / 16-23; lanes 16-31 K 8-15 / 24-31.
template <typename PtrT>
__device__ __forceinline__ v16h load_frag(PtrT base_row, int c, int half) {
  V16U t;
  t.h[0] = *(const v8h*)(base_row + 32 * c + 8 * half);
  t.h[1] = *(const v8h*)(base_row + 32 * c + 16 + 8 * half);
  return t.v;
}

// ---------------------------------------------------------------------------
// Kernel 2: fused Gram-panel WMMA + fixed-shift masked logsumexp + pos sums.
// Block = 4 waves = 64 rows; 32-column B panels staged to LDS by the TDM
// (double-buffered, overlapped with compute); two 16-col WMMA sub-tiles per
// stage.  Per-lane accumulators, shuffle-add merge at the end.
// ---------------------------------------------------------------------------
__global__ __launch_bounds__(32 * WAVES_PB) void supcon_main(
    const _Float16* __restrict__ xh, const int* __restrict__ labels,
    float* __restrict__ partials, int N) {
  __shared__ alignas(16) unsigned char panel[2][PANEL_BYTES];

  const int lane = threadIdx.x & 31;
  const int wave = threadIdx.x >> 5;
  const int tile = blockIdx.x * WAVES_PB + wave;  // global 16-row tile id
  const int r0   = tile * 16;
  const int half = lane >> 4;  // C slots: 0 -> rows r..r+7, 1 -> rows r+8..r+15
  const int lmod = lane & 15;  // N column within tile / fragment row id

  // A fragments: this wave's 16 rows x 128 K, resident for the whole loop.
  v16h a[4];
  {
    const _Float16* rowp = xh + (size_t)(r0 + lmod) * D_DIM;
#pragma unroll
    for (int c = 0; c < 4; ++c) a[c] = load_frag(rowp, c, half);
  }

  const int rbase = r0 + 8 * half;  // slot r -> matrix row rbase + r
  int lrow[8];
#pragma unroll
  for (int r = 0; r < 8; ++r) lrow[r] = labels[rbase + r];

  float l[8], p[8], cnt[8];
#pragma unroll
  for (int r = 0; r < 8; ++r) { l[r] = 0.0f; p[r] = 0.0f; cnt[r] = 0.0f; }

  const unsigned long long gbase = (unsigned long long)(uintptr_t)xh;
  const unsigned ldsbase = (unsigned)(uintptr_t)(void*)&panel[0][0];
  const int nstages = N / STAGE_ROWS;

#if HAVE_TDM
  // D# group 1 (constant): data_size=8B, pad 4 DWORDs every 64 DWORDs,
  // 2D tile 32x32 elements (32 rows x 256B), contiguous rows (stride 32 elem).
  const unsigned gw0 = (3u << 16) | (1u << 20) | (5u << 22) | (3u << 25);
  const int32x8 g1 = {(int)gw0, (int)(32u << 16), (int)(32u << 16),
                      (int)(32u << 16), STAGE_ROWS, 32, 0, 0};
  const int32x4 z4 = {0, 0, 0, 0};
#if __clang_major__ >= 23
  const int32x8 z8 = {0, 0, 0, 0, 0, 0, 0, 0};
#endif
#endif

  // Stage 32 Gram columns into LDS buffer `buf` (TDM on wave 0, else copy).
  auto stage = [&](int j, int buf) {
#if HAVE_TDM
    if (wave == 0) {
      const unsigned long long ga =
          gbase + (unsigned long long)j * STAGE_ROWS * ROW_BYTES;
      const unsigned lds = ldsbase + (unsigned)buf * PANEL_BYTES;
      const uint32x4 g0 = {1u, lds, (unsigned)ga,
                           (unsigned)(ga >> 32) | 0x80000000u};
#if __clang_major__ >= 23
      __builtin_amdgcn_tensor_load_to_lds(g0, g1, z4, z4, z8, 0);
#else
      __builtin_amdgcn_tensor_load_to_lds(g0, g1, z4, z4, 0);
#endif
    }
#else
    const unsigned char* src =
        (const unsigned char*)xh + (size_t)j * STAGE_ROWS * ROW_BYTES;
    for (int t = threadIdx.x; t < STAGE_ROWS * 16; t += 32 * WAVES_PB) {
      const int row = t >> 4, q = t & 15;
      *(v8h*)(&panel[buf][row * LDS_ROW_B + q * 16]) =
          *(const v8h*)(src + row * ROW_BYTES + q * 16);
    }
#endif
  };
  auto stage_wait = [&]() {
#if HAVE_TDM
    if (wave == 0) __builtin_amdgcn_s_wait_tensorcnt(0);
#endif
    __syncthreads();
  };

  stage(0, 0);
  stage_wait();

  for (int j = 0; j < nstages; ++j) {
    const int cur = j & 1;
    if (j + 1 < nstages) stage(j + 1, cur ^ 1);  // overlap DMA with compute

#pragma unroll
    for (int t = 0; t < 2; ++t) {  // two 16-column sub-tiles per stage
      const _Float16* bp =
          (const _Float16*)(&panel[cur][(t * 16 + lmod) * LDS_ROW_B]);
      // Load all 4 B chunks first so the 8 ds_load_b128 issue back-to-back
      // and the WMMA chain only waits incrementally.
      v16h b[4];
#pragma unroll
      for (int c = 0; c < 4; ++c) b[c] = load_frag(bp, c, half);

      v8f acc = {};
#pragma unroll
      for (int c = 0; c < 4; ++c)
        acc = __builtin_amdgcn_wmma_f32_16x16x32_f16(
            false, a[c], false, b[c], (short)0, acc, false, false);

      const int col = j * STAGE_ROWS + t * 16 + lmod;
      const int lc  = labels[col];
      const int dr  = col - rbase;  // diag in slot r <=> dr == r
#pragma unroll
      for (int r = 0; r < 8; ++r) {
        const float s    = acc[r] * INV_TEMP;
        const float e    = __expf(s - SHIFT);  // logsumexp shift-invariant
        const bool  diag = (dr == r);
        const bool  pos  = (lc == lrow[r]) && !diag;
        l[r]   += diag ? 0.0f : e;
        p[r]   += pos ? s : 0.0f;
        cnt[r] += pos ? 1.0f : 0.0f;
      }
    }

    stage_wait();
  }

  // Merge the 16 per-lane column partials of each row (plain sums).
  float lsum = 0.0f;
#pragma unroll
  for (int r = 0; r < 8; ++r) {
    float ll = l[r], pp = p[r], cc = cnt[r];
#pragma unroll
    for (int mask = 1; mask < 16; mask <<= 1) {
      ll += __shfl_xor(ll, mask, 32);
      pp += __shfl_xor(pp, mask, 32);
      cc += __shfl_xor(cc, mask, 32);
    }
    // loss_row = -(pp - c*SHIFT - c*log(ll)) / (c + tiny)
    lsum += (cc * (SHIFT + __logf(ll)) - pp) / (cc + SMALL_VAL);
  }
  const float other = __shfl_xor(lsum, 16, 32);  // combine the two row halves
  if (lane == 0) partials[tile] = lsum + other;
}

// ---------------------------------------------------------------------------
// Kernel 3: deterministic fixed-order tree reduction of per-wave partials.
// ---------------------------------------------------------------------------
__global__ __launch_bounds__(512) void supcon_reduce(
    const float* __restrict__ partials, float* __restrict__ out, int n) {
  __shared__ float sm[512];
  const int t = threadIdx.x;
  float acc = 0.0f;
  for (int i = t; i < n; i += 512) acc += partials[i];
  sm[t] = acc;
  __syncthreads();
#pragma unroll
  for (int s = 256; s > 0; s >>= 1) {
    if (t < s) sm[t] += sm[t + s];
    __syncthreads();
  }
  if (t == 0) out[0] = sm[0];
}

extern "C" void kernel_launch(void* const* d_in, const int* in_sizes, int n_in,
                              void* d_out, int out_size, void* d_ws, size_t ws_size,
                              hipStream_t stream) {
  const float* emb    = (const float*)d_in[0];
  const int*   labels = (const int*)d_in[1];
  float*       out    = (float*)d_out;
  const int N = in_sizes[0] / D_DIM;  // 8192

  _Float16* xh       = (_Float16*)d_ws;
  float*    partials = (float*)((char*)d_ws + (size_t)N * D_DIM * sizeof(_Float16));
  const int ntiles = N / 16;  // 512

  supcon_normalize<<<N / 8, 256, 0, stream>>>(emb, xh, N);
  supcon_main<<<ntiles / WAVES_PB, 32 * WAVES_PB, 0, stream>>>(xh, labels,
                                                               partials, N);
  supcon_reduce<<<1, 512, 0, stream>>>(partials, out, ntiles);
}
